// ReprogrammingLayer_46050639348104
// MI455X (gfx1250) — matrix operand order repaired and checked by reading, not compiled, not run on hardware
//
#include <hip/hip_runtime.h>

// ---------------------------------------------------------------------------
// ReprogrammingLayer (cross attention) for MI455X / gfx1250, wave32 + WMMA.
//   1) f32 -> bf16 conversion of activations; f32 -> bf16 transposed weights
//   2) bf16 GEMMs (v_wmma_f32_16x16x32_bf16), async-LDS double-buffered tiles
//      (BK=64: one barrier per 16 WMMAs)
//   3) flash-attention kernel (online softmax, async-LDS staged K/V, WMMA)
//   4) bf16 GEMM with f32 epilogue for the output projection
// CDNA5 specifics: global_load_async_to_lds_b128 (ASYNCcnt), s_wait_asynccnt,
// s_wait_dscnt, v_wmma_f32_16x16x32_bf16.
// ---------------------------------------------------------------------------

typedef __attribute__((ext_vector_type(16))) __bf16 v16bf;
typedef __attribute__((ext_vector_type(8)))  __bf16 bf16x8;
typedef __attribute__((ext_vector_type(8)))  float  v8f;

#define BB 8
#define LL 1024
#define SS 1000
#define SPAD 1024
#define DM 1024
#define DL 4096
#define HH 16
#define EE 64
#define HE 1024   // H*E

static __device__ inline v8f wmma_bf16(v16bf a, v16bf b, v8f c) {
  return __builtin_amdgcn_wmma_f32_16x16x32_bf16(
      /*neg_a=*/false, a, /*neg_b=*/false, b,
      /*c_mod=*/(short)0, c, /*reuse_a=*/false, /*reuse_b=*/false);
}

// Async DMA: global -> LDS, 16B per lane, tracked by ASYNCcnt.
// LDS destination address = low 32 bits of the generic pointer (aperture in
// the high half), which is the wave-relative LDS byte offset.
static __device__ inline void async_ld_b128(void* lds_dst, const void* gsrc) {
  unsigned l = (unsigned)(unsigned long long)lds_dst;
  asm volatile("global_load_async_to_lds_b128 %0, %1, off"
               :: "v"(l), "v"(gsrc) : "memory");
}
static __device__ inline void wait_async0() {
  asm volatile("s_wait_asynccnt 0x0" ::: "memory");
}

// A-operand fragment (16x32, MxK): lanes 0-15 row=ln hold K [koff..koff+7] and
// [koff+16..koff+23]; lanes 16-31 are +8 in K.  (ISA 7.12.2, 16-bit A layout)
static __device__ inline v16bf frag_a(const __bf16* base, int row, int stride,
                                      int koff, int asel) {
  const __bf16* p = base + row * stride + koff + asel;
  union { v16bf v; bf16x8 h[2]; } u;
  u.h[0] = *(const bf16x8*)(p);
  u.h[1] = *(const bf16x8*)(p + 16);
  return u.v;
}

// B-operand fragment (32x16, KxN) from transposed storage Bt[n][k]:
// lanes 0-15 (col=ln) hold K [koff..koff+7] and [koff+16..koff+23]; lanes 16-31 are +8 in K.
static __device__ inline v16bf frag_b(const __bf16* base, int col, int stride,
                                      int koff, int bsel) {
  const __bf16* p = base + col * stride + koff + bsel;
  union { v16bf v; bf16x8 h[2]; } u;
  u.h[0] = *(const bf16x8*)(p);
  u.h[1] = *(const bf16x8*)(p + 16);
  return u.v;
}

// ---------------------------------------------------------------------------
// f32 -> bf16, vectorized by 8 (all element counts are multiples of 8)
// ---------------------------------------------------------------------------
__global__ void cvt_f32_bf16(const float* __restrict__ in,
                             __bf16* __restrict__ out, long n) {
  long i = (long)blockIdx.x * blockDim.x + threadIdx.x;
  if (i * 8 >= n) return;
  float4 a = ((const float4*)in)[2 * i];
  float4 b = ((const float4*)in)[2 * i + 1];
  bf16x8 o;
  o[0] = (__bf16)a.x; o[1] = (__bf16)a.y; o[2] = (__bf16)a.z; o[3] = (__bf16)a.w;
  o[4] = (__bf16)b.x; o[5] = (__bf16)b.y; o[6] = (__bf16)b.z; o[7] = (__bf16)b.w;
  ((bf16x8*)out)[i] = o;
}

// f32 [K][N] -> bf16 transposed [N][K]
__global__ void cvt_f32_bf16_tr(const float* __restrict__ W,
                                __bf16* __restrict__ Wt, int K, int N) {
  long i = (long)blockIdx.x * blockDim.x + threadIdx.x;
  if (i >= (long)K * N) return;
  int n = (int)(i % N);
  int k = (int)(i / N);
  Wt[(long)n * K + k] = (__bf16)W[i];
}

// ---------------------------------------------------------------------------
// bf16 GEMM: C[M][N] = A[M][K] * B[K][N] + bias, with B given transposed
// (Bt[N][K]).  128x128 block tile, BK=64, 8 waves, each wave 32x64.
// Tiles staged with async global->LDS DMA, double-buffered: one barrier per
// 64-deep K-tile; next tile's DMA overlaps current tile's 16 WMMAs.
// EPI: 0 = bf16 row-major, 1 = bf16 K-layout [h][s][e],
//      2 = bf16 Vt-layout [h][e][s], 3 = f32 row-major.
// Rows m >= M are clamped to row M-1 (garbage is masked downstream).
// ---------------------------------------------------------------------------
template <int EPI>
__global__ __launch_bounds__(256) void gemm_bf16(
    const __bf16* __restrict__ A, const __bf16* __restrict__ Bt,
    const float* __restrict__ bias, void* __restrict__ Cout,
    int M, int N, int K) {
  __shared__ __bf16 As[2][128 * 64];   // 16KB per buffer
  __shared__ __bf16 Bs[2][128 * 64];   // 16KB per buffer

  const int tid  = threadIdx.x;
  const int lane = tid & 31;
  const int wid  = tid >> 5;
  const int ln   = lane & 15;
  const int asel = (lane & 16) ? 8 : 0;
  const int bsel = (lane & 16) ? 8 : 0;
  const int crow = (lane & 16) ? 8 : 0;
  const int waveM = wid >> 1;      // 0..3  -> 32 rows each
  const int waveN = wid & 1;       // 0..1  -> 64 cols each
  const int m0 = blockIdx.x * 128;
  const int n0 = blockIdx.y * 128;

  // staging: 128 rows x 64 cols = 1024 x 16B chunks per matrix;
  // each thread DMAs 4 chunks for A and 4 for B per K-tile
  int srow[4], sseg[4], garow[4];
#pragma unroll
  for (int p = 0; p < 4; ++p) {
    int c = tid + p * 256;           // 0..1023
    srow[p] = c >> 3;                // row 0..127
    sseg[p] = (c & 7) * 8;           // col segment (8 bf16 = 16B)
    garow[p] = min(m0 + srow[p], M - 1);
  }

  v8f acc[2][4] = {};

  const int nk = K / 64;
  // prologue: stage K-tile 0 into buffer 0
#pragma unroll
  for (int p = 0; p < 4; ++p) {
    async_ld_b128(&As[0][srow[p] * 64 + sseg[p]],
                  A + (long)garow[p] * K + sseg[p]);
    async_ld_b128(&Bs[0][srow[p] * 64 + sseg[p]],
                  Bt + (long)(n0 + srow[p]) * K + sseg[p]);
  }

  for (int i = 0; i < nk; ++i) {
    wait_async0();        // our DMA into buf (i&1) landed
    __syncthreads();      // everyone's did
    if (i + 1 < nk) {     // prefetch next K-tile into the other buffer
      const int nb = (i + 1) & 1, kt = (i + 1) * 64;
#pragma unroll
      for (int p = 0; p < 4; ++p) {
        async_ld_b128(&As[nb][srow[p] * 64 + sseg[p]],
                      A + (long)garow[p] * K + kt + sseg[p]);
        async_ld_b128(&Bs[nb][srow[p] * 64 + sseg[p]],
                      Bt + (long)(n0 + srow[p]) * K + kt + sseg[p]);
      }
    }
    const __bf16* Ab = As[i & 1];
    const __bf16* Bb = Bs[i & 1];
#pragma unroll
    for (int ks = 0; ks < 2; ++ks) {           // two 32-deep WMMA steps
      const int ko = ks * 32;
      v16bf af[2], bf[4];
#pragma unroll
      for (int mt = 0; mt < 2; ++mt)
        af[mt] = frag_a(Ab, waveM * 32 + mt * 16 + ln, 64, ko, asel);
#pragma unroll
      for (int nt = 0; nt < 4; ++nt)
        bf[nt] = frag_b(Bb, waveN * 64 + nt * 16 + ln, 64, ko, bsel);
#pragma unroll
      for (int mt = 0; mt < 2; ++mt)
#pragma unroll
        for (int nt = 0; nt < 4; ++nt)
          acc[mt][nt] = wmma_bf16(af[mt], bf[nt], acc[mt][nt]);
    }
  }

  // epilogue
#pragma unroll
  for (int mt = 0; mt < 2; ++mt) {
#pragma unroll
    for (int nt = 0; nt < 4; ++nt) {
      int mb = m0 + waveM * 32 + mt * 16;
      int nb = n0 + waveN * 64 + nt * 16;
      int n  = nb + ln;
      float bv = bias ? bias[n] : 0.0f;
#pragma unroll
      for (int j = 0; j < 8; ++j) {
        int m   = mb + j + crow;
        float v = acc[mt][nt][j] + bv;
        if (EPI == 0) {
          ((__bf16*)Cout)[(long)m * N + n] = (__bf16)v;
        } else if (EPI == 1) {        // K layout: [h][s][e], s padded to SPAD
          ((__bf16*)Cout)[(long)(n >> 6) * SPAD * EE + (long)m * EE + (n & 63)] =
              (__bf16)v;
        } else if (EPI == 2) {        // Vt layout: [h][e][s]
          ((__bf16*)Cout)[(long)(n >> 6) * EE * SPAD + (long)(n & 63) * SPAD + m] =
              (__bf16)v;
        } else {                      // f32 row-major (final output)
          ((float*)Cout)[(long)m * N + n] = v;
        }
      }
    }
  }
}

// ---------------------------------------------------------------------------
// Flash attention: grid (L/128, H, B), 8 waves x 16 query rows each.
// K slab [128][64] and V slab [64][128] async-DMAed into double-buffered LDS;
// online softmax in registers; prob tile round-trips LDS to become A-operand.
// ---------------------------------------------------------------------------
__global__ __launch_bounds__(256) void attn_kernel(
    const __bf16* __restrict__ Q,   // [B*L][HE]
    const __bf16* __restrict__ Kb,  // [H][SPAD][E]
    const __bf16* __restrict__ Vt,  // [H][E][SPAD]
    __bf16* __restrict__ O) {       // [B*L][HE]
  __shared__ __bf16 Ksh[2][128 * 64];     // [s][e]   16KB each
  __shared__ __bf16 Vsh[2][64 * 128];     // [e][s]   16KB each
  __shared__ __bf16 Psh[8 * 16 * 64];     // per-wave prob tile [16][64]

  const int tid  = threadIdx.x;
  const int lane = tid & 31;
  const int wid  = tid >> 5;
  const int ln   = lane & 15;
  const int asel = (lane & 16) ? 8 : 0;
  const int bsel = (lane & 16) ? 8 : 0;
  const int crow = (lane & 16) ? 8 : 0;

  const int b  = blockIdx.z;
  const int h  = blockIdx.y;
  const int m0 = blockIdx.x * 128 + wid * 16;

  const __bf16* Qrow  = Q + ((long)(b * LL) + m0 + ln) * HE + h * EE;
  const __bf16* Khead = Kb + (long)h * SPAD * EE;
  const __bf16* Vhead = Vt + (long)h * EE * SPAD;
  __bf16* Pw = Psh + wid * 16 * 64;

  // Q fragments (16 rows x 64 e), kept resident
  v16bf qf[2];
#pragma unroll
  for (int half = 0; half < 2; ++half) {
    union { v16bf v; bf16x8 h2[2]; } u;
    const __bf16* p = Qrow + half * 32 + asel;
    u.h2[0] = *(const bf16x8*)(p);
    u.h2[1] = *(const bf16x8*)(p + 16);
    qf[half] = u.v;
  }

  v8f oacc[4] = {};
  float mrun[8], lrun[8];
#pragma unroll
  for (int j = 0; j < 8; ++j) { mrun[j] = -1e30f; lrun[j] = 0.0f; }

  // prologue: stage chunk 0 into buffer 0
#pragma unroll
  for (int p = 0; p < 4; ++p) {
    int c = tid + p * 256;                    // 0..1023 chunks of 8 elems
    async_ld_b128(&Ksh[0][c * 8], Khead + c * 8);
    int e = c >> 4, seg = (c & 15) * 8;
    async_ld_b128(&Vsh[0][e * 128 + seg], Vhead + (long)e * SPAD + seg);
  }

  for (int i = 0; i < SPAD / 128; ++i) {
    wait_async0();
    __syncthreads();
    if (i + 1 < SPAD / 128) {                 // prefetch next chunk
      const int nb = (i + 1) & 1, s0n = (i + 1) * 128;
#pragma unroll
      for (int p = 0; p < 4; ++p) {
        int c = tid + p * 256;
        async_ld_b128(&Ksh[nb][c * 8], Khead + (long)s0n * EE + c * 8);
        int e = c >> 4, seg = (c & 15) * 8;
        async_ld_b128(&Vsh[nb][e * 128 + seg],
                      Vhead + (long)e * SPAD + s0n + seg);
      }
    }
    const __bf16* Kc = Ksh[i & 1];
    const __bf16* Vc = Vsh[i & 1];
    const int s0 = i * 128;

#pragma unroll
    for (int g = 0; g < 2; ++g) {             // two 64-key groups per chunk
      const int ng = g * 64;
      v8f sc[4];
#pragma unroll
      for (int t = 0; t < 4; ++t) {
        v16bf kf0 = frag_b(Kc, ng + t * 16 + ln, 64, 0,  bsel);
        v16bf kf1 = frag_b(Kc, ng + t * 16 + ln, 64, 32, bsel);
        v8f a = {};
        a = wmma_bf16(qf[0], kf0, a);
        a = wmma_bf16(qf[1], kf1, a);
        const int sg = s0 + ng + t * 16 + ln;
        const bool valid = (sg < SS);
#pragma unroll
        for (int j = 0; j < 8; ++j)
          sc[t][j] = valid ? a[j] * 0.125f : -1e30f;   // 1/sqrt(64), mask pad
      }

      float alpha[8];
#pragma unroll
      for (int j = 0; j < 8; ++j) {
        float r = fmaxf(fmaxf(sc[0][j], sc[1][j]), fmaxf(sc[2][j], sc[3][j]));
        r = fmaxf(r, __shfl_xor(r, 1, 32));
        r = fmaxf(r, __shfl_xor(r, 2, 32));
        r = fmaxf(r, __shfl_xor(r, 4, 32));
        r = fmaxf(r, __shfl_xor(r, 8, 32));
        float mn = fmaxf(mrun[j], r);
        alpha[j] = __expf(mrun[j] - mn);
        mrun[j]  = mn;
      }
#pragma unroll
      for (int j = 0; j < 8; ++j) {
        float p0 = __expf(sc[0][j] - mrun[j]);
        float p1 = __expf(sc[1][j] - mrun[j]);
        float p2 = __expf(sc[2][j] - mrun[j]);
        float p3 = __expf(sc[3][j] - mrun[j]);
        sc[0][j] = p0; sc[1][j] = p1; sc[2][j] = p2; sc[3][j] = p3;
        float s = p0 + p1 + p2 + p3;
        s += __shfl_xor(s, 1, 32);
        s += __shfl_xor(s, 2, 32);
        s += __shfl_xor(s, 4, 32);
        s += __shfl_xor(s, 8, 32);
        lrun[j] = lrun[j] * alpha[j] + s;
      }
      // rescale running output, spill prob tile to LDS (C-layout -> A-layout)
#pragma unroll
      for (int t = 0; t < 4; ++t)
#pragma unroll
        for (int j = 0; j < 8; ++j) {
          oacc[t][j] *= alpha[j];
          Pw[(j + crow) * 64 + t * 16 + ln] = (__bf16)sc[t][j];
        }
      asm volatile("s_wait_dscnt 0" ::: "memory");  // wave-local LDS RAW

      v16bf pf0 = frag_a(Pw, ln, 64, 0,  asel);
      v16bf pf1 = frag_a(Pw, ln, 64, 32, asel);
#pragma unroll
      for (int t = 0; t < 4; ++t) {
        v16bf vf0 = frag_b(Vc, t * 16 + ln, 128, ng,      bsel);
        v16bf vf1 = frag_b(Vc, t * 16 + ln, 128, ng + 32, bsel);
        oacc[t] = wmma_bf16(pf0, vf0, oacc[t]);
        oacc[t] = wmma_bf16(pf1, vf1, oacc[t]);
      }
    }
  }

  float inv[8];
#pragma unroll
  for (int j = 0; j < 8; ++j) inv[j] = 1.0f / lrun[j];
#pragma unroll
  for (int t = 0; t < 4; ++t)
#pragma unroll
    for (int j = 0; j < 8; ++j) {
      int m = m0 + j + crow;
      int e = t * 16 + ln;
      O[((long)(b * LL) + m) * HE + h * EE + e] = (__bf16)(oacc[t][j] * inv[j]);
    }
}

// ---------------------------------------------------------------------------
// Host launcher.  Workspace layout (bf16, all offsets 256B-aligned), ~94 MiB.
// ---------------------------------------------------------------------------
extern "C" void kernel_launch(void* const* d_in, const int* in_sizes, int n_in,
                              void* d_out, int out_size, void* d_ws,
                              size_t ws_size, hipStream_t stream) {
  const float* tgt = (const float*)d_in[0];
  const float* src = (const float*)d_in[1];
  const float* val = (const float*)d_in[2];
  const float* Wq  = (const float*)d_in[3];
  const float* bq  = (const float*)d_in[4];
  const float* Wk  = (const float*)d_in[5];
  const float* bk  = (const float*)d_in[6];
  const float* Wv  = (const float*)d_in[7];
  const float* bv  = (const float*)d_in[8];
  const float* Wo  = (const float*)d_in[9];
  const float* bo  = (const float*)d_in[10];

  char* ws = (char*)d_ws;
  size_t off = 0;
  auto alloc = [&](size_t bytes) { char* p = ws + off; off += bytes; return p; };
  __bf16* WqT = (__bf16*)alloc((size_t)HE * DM * 2);        //  2 MiB  [N=HE][K=DM]
  __bf16* WkT = (__bf16*)alloc((size_t)HE * DL * 2);        //  8 MiB
  __bf16* WvT = (__bf16*)alloc((size_t)HE * DL * 2);        //  8 MiB
  __bf16* WoT = (__bf16*)alloc((size_t)DL * HE * 2);        //  8 MiB  [N=DL][K=HE]
  __bf16* Xq  = (__bf16*)alloc((size_t)BB * LL * DM * 2);   // 16 MiB
  __bf16* Xs  = (__bf16*)alloc((size_t)SS * DL * 2);        //  8 MiB
  __bf16* Xv  = (__bf16*)alloc((size_t)SS * DL * 2);        //  8 MiB
  __bf16* Qb  = (__bf16*)alloc((size_t)BB * LL * HE * 2);   // 16 MiB
  __bf16* Kbf = (__bf16*)alloc((size_t)HH * SPAD * EE * 2); //  2 MiB
  __bf16* Vtb = (__bf16*)alloc((size_t)HH * EE * SPAD * 2); //  2 MiB
  __bf16* Ob  = (__bf16*)alloc((size_t)BB * LL * HE * 2);   // 16 MiB
  (void)ws_size;

  // 1) conversions
  {
    long n1 = (long)BB * LL * DM;   // 8388608
    cvt_f32_bf16<<<(int)(n1 / 8 / 256), 256, 0, stream>>>(tgt, Xq, n1);
    long n2 = (long)SS * DL;        // 4096000
    cvt_f32_bf16<<<(int)(n2 / 8 / 256), 256, 0, stream>>>(src, Xs, n2);
    cvt_f32_bf16<<<(int)(n2 / 8 / 256), 256, 0, stream>>>(val, Xv, n2);
    cvt_f32_bf16_tr<<<(DM * HE + 255) / 256, 256, 0, stream>>>(Wq, WqT, DM, HE);
    cvt_f32_bf16_tr<<<(DL * HE + 255) / 256, 256, 0, stream>>>(Wk, WkT, DL, HE);
    cvt_f32_bf16_tr<<<(DL * HE + 255) / 256, 256, 0, stream>>>(Wv, WvT, DL, HE);
    cvt_f32_bf16_tr<<<(HE * DL + 255) / 256, 256, 0, stream>>>(Wo, WoT, HE, DL);
  }

  // 2) projections
  gemm_bf16<0><<<dim3(BB * LL / 128, HE / 128), 256, 0, stream>>>(
      Xq, WqT, bq, Qb, BB * LL, HE, DM);
  gemm_bf16<1><<<dim3(SPAD / 128, HE / 128), 256, 0, stream>>>(
      Xs, WkT, bk, Kbf, SS, HE, DL);             // pad rows masked in attention
  gemm_bf16<2><<<dim3(SPAD / 128, HE / 128), 256, 0, stream>>>(
      Xv, WvT, bv, Vtb, SS, HE, DL);             // pad cols weighted by exactly 0

  // 3) attention
  attn_kernel<<<dim3(LL / 128, HH, BB), 256, 0, stream>>>(Qb, Kbf, Vtb, Ob);

  // 4) output projection (f32 out)
  gemm_bf16<3><<<dim3(BB * LL / 128, DL / 128), 256, 0, stream>>>(
      Ob, WoT, bo, (float*)d_out, BB * LL, DL, HE);
}
